// ObjectDetector_70738111365953
// MI455X (gfx1250) — matrix operand order repaired and checked by reading, not compiled, not run on hardware
//
#include <hip/hip_runtime.h>
#include <hip/hip_bf16.h>
#include <math.h>
#include <stdint.h>

typedef _Float16 v16h __attribute__((ext_vector_type(16)));
typedef _Float16 v8h  __attribute__((ext_vector_type(8)));
typedef float    v8f  __attribute__((ext_vector_type(8)));
typedef int      v4i  __attribute__((ext_vector_type(4)));
typedef __attribute__((address_space(1))) v4i* gptr4;   // global int4*
typedef __attribute__((address_space(3))) v4i* lptr4;   // LDS int4*

// Problem constants (from reference setup_inputs)
#define BATCH   8
#define NBOX    300
#define IMH     512
#define IMW     512
#define CROP    64
#define TOPK    20
#define COUT    256
#define FDIM    4096
#define KCONV   147          // 3*7*7
#define KPAD    160          // padded to 5 * 32
#define NCROPS  (BATCH*TOPK) // 160
#define MROWS   176          // 8 fullframe + 160 crops, padded to 11*16
#define MTILE   64           // output pixels per conv block

// ---------------------------------------------------------------- normalize
__global__ void k_normalize(const float* __restrict__ x, float* __restrict__ img, int total) {
    int i = blockIdx.x * 256 + threadIdx.x;
    if (i >= total) return;
    int c = (i / (IMH * IMW)) % 3;
    const float mean[3] = {0.485f, 0.456f, 0.406f};
    const float stdv[3] = {0.229f, 0.224f, 0.225f};
    img[i] = (x[i] - mean[c]) / stdv[c];
}

// ------------------------------------------------- conv weights -> f16 [N][KPAD]
__global__ void k_prep_wc(const float* __restrict__ Wc, _Float16* __restrict__ Wt) {
    int i = blockIdx.x * 256 + threadIdx.x;      // COUT*KPAD
    if (i >= COUT * KPAD) return;
    int n = i / KPAD, k = i % KPAD;
    float v = (k < KCONV) ? Wc[n * KCONV + k] : 0.f;  // Wc is [256][3][7][7] contiguous
    Wt[i] = (_Float16)v;
}

// --------------------------------------------- FC weights -> f16 [N=4096][K=256]
__global__ void k_prep_wf(const float* __restrict__ Wf, _Float16* __restrict__ WfH) {
    int i = blockIdx.x * 256 + threadIdx.x;      // FDIM*COUT
    if (i >= FDIM * COUT) return;
    int n = i / COUT, k = i % COUT;
    WfH[i] = (_Float16)Wf[k * FDIM + n];         // Wf is [256][4096]
}

__global__ void k_zero(float* __restrict__ p, int total) {
    int i = blockIdx.x * 256 + threadIdx.x;
    if (i < total) p[i] = 0.f;
}

// ---------------------------------------------------------------- top-k select
__global__ void k_topk(const float* __restrict__ boxes, const float* __restrict__ scores,
                       const int* __restrict__ labels, float* __restrict__ objOut,
                       float* __restrict__ validOut, float* __restrict__ maskf) {
    __shared__ float ms[NBOX];
    __shared__ float rv[256];
    __shared__ int   ri[256];
    int b = blockIdx.x, tid = threadIdx.x;
    for (int i = tid; i < NBOX; i += 256) {
        int lab = labels[b * NBOX + i];
        bool cls = (lab == 2) | (lab == 3) | (lab == 4) | (lab == 6) | (lab == 8);
        float sc = scores[b * NBOX + i];
        ms[i] = (cls && sc > 0.8f) ? sc : -INFINITY;
    }
    __syncthreads();
    for (int j = 0; j < TOPK; j++) {
        float bv = -INFINITY; int bi = NBOX;
        for (int i = tid; i < NBOX; i += 256) {
            float v = ms[i];
            if (v > bv || (v == bv && i < bi)) { bv = v; bi = i; }
        }
        rv[tid] = bv; ri[tid] = bi;
        __syncthreads();
        for (int s = 128; s > 0; s >>= 1) {
            if (tid < s) {
                float v2 = rv[tid + s]; int i2 = ri[tid + s];
                if (v2 > rv[tid] || (v2 == rv[tid] && i2 < ri[tid])) { rv[tid] = v2; ri[tid] = i2; }
            }
            __syncthreads();
        }
        if (tid == 0) {
            float bestv = rv[0]; int besti = ri[0];
            float x1 = 0.f, y1 = 0.f, x2 = 0.f, y2 = 0.f, val = 0.f;
            if (bestv > -INFINITY) {
                val = 1.f;
                x1 = boxes[(b * NBOX + besti) * 4 + 0];
                y1 = boxes[(b * NBOX + besti) * 4 + 1];
                x2 = boxes[(b * NBOX + besti) * 4 + 2];
                y2 = boxes[(b * NBOX + besti) * 4 + 3];
                ms[besti] = -INFINITY;
            }
            int q = b * TOPK + j;
            objOut[q * 4 + 0] = x1; objOut[q * 4 + 1] = y1;
            objOut[q * 4 + 2] = x2; objOut[q * 4 + 3] = y2;
            validOut[q] = val;
            bool deg = (x2 <= x1) || (y2 <= y1);
            maskf[q] = (val > 0.f && !deg) ? 1.f : 0.f;
        }
        __syncthreads();
    }
}

// ---------------------------------------------------------------- RoIAlign
__global__ void k_roi(const float* __restrict__ img, const float* __restrict__ objOut,
                      float* __restrict__ crops) {
    int g = blockIdx.x * 256 + threadIdx.x;
    if (g >= NCROPS * 3 * CROP * CROP) return;
    int q   = g / (3 * CROP * CROP);
    int rem = g % (3 * CROP * CROP);
    int c = rem / (CROP * CROP), p = rem % (CROP * CROP);
    int oy = p >> 6, ox = p & 63;
    int b = q / TOPK;
    float x1 = objOut[q * 4 + 0], y1 = objOut[q * 4 + 1];
    float x2 = objOut[q * 4 + 2], y2 = objOut[q * 4 + 3];
    float ty = (oy + 0.5f) * (1.f / CROP), tx = (ox + 0.5f) * (1.f / CROP);
    float ys = y1 + ty * (y2 - y1), xs = x1 + tx * (x2 - x1);
    float y0f = floorf(ys), x0f = floorf(xs);
    float wy = ys - y0f, wx = xs - x0f;
    int y0 = min(max((int)y0f, 0), IMH - 1); int y1i = min(y0 + 1, IMH - 1);
    int x0 = min(max((int)x0f, 0), IMW - 1); int x1i = min(x0 + 1, IMW - 1);
    const float* base = img + ((size_t)b * 3 + c) * IMH * IMW;
    float Ia = base[y0 * IMW + x0],  Ib = base[y0 * IMW + x1i];
    float Ic = base[y1i * IMW + x0], Id = base[y1i * IMW + x1i];
    crops[g] = Ia * (1.f - wy) * (1.f - wx) + Ib * (1.f - wy) * wx
             + Ic * wy * (1.f - wx)         + Id * wy * wx;
}

// ------------------------------------- WMMA im2col conv + ReLU + mean pool
// block = 256 threads (8 waves). Tile: M=64 output pixels x N=256 channels.
// Wt staged into LDS once per block (async-to-LDS if available).
// wave w handles N-tiles {2w,2w+1} x 4 M-tiles; K loop: 5 steps of 32.
__global__ void k_convpool(const float* __restrict__ src, const _Float16* __restrict__ Wt,
                           const float* __restrict__ bc, float* __restrict__ pooled,
                           int nPixTiles, int H, int W, int OW, float invN, int pooledBase) {
    __shared__ __align__(16) _Float16 sA[MTILE * KPAD];   // 20 KB im2col patches
    __shared__ __align__(16) _Float16 sB[COUT * KPAD];    // 80 KB weights
    int img = blockIdx.x / nPixTiles;
    int pt  = blockIdx.x % nPixTiles;
    int tid = threadIdx.x;
    const float* im = src + (size_t)img * 3 * H * W;

    // ---- stage full weight matrix into LDS (16B chunks per lane)
#if __has_builtin(__builtin_amdgcn_global_load_async_to_lds_b128)
    for (int e = tid; e < (COUT * KPAD) / 8; e += 256) {
        __builtin_amdgcn_global_load_async_to_lds_b128(
            (gptr4)(uintptr_t)(Wt + e * 8),
            (lptr4)(sB + e * 8),
            0, 0);
    }
#if __has_builtin(__builtin_amdgcn_s_wait_asynccnt)
    __builtin_amdgcn_s_wait_asynccnt(0);
#else
    asm volatile("s_wait_asynccnt 0" ::: "memory");
#endif
#else
    for (int e = tid; e < (COUT * KPAD) / 8; e += 256)
        ((float4*)sB)[e] = ((const float4*)Wt)[e];
#endif

    // ---- im2col: patch tile [64 pixels][160 k] into LDS as f16 (SAME pad_lo = 1)
    for (int e = tid; e < MTILE * KPAD; e += 256) {
        int m = e / KPAD, k = e % KPAD;
        int p = pt * MTILE + m;
        int oy = p / OW, ox = p % OW;
        float v = 0.f;
        if (k < KCONV) {
            int c = k / 49, r = k % 49, ky = r / 7, kx = r % 7;
            int iy = oy * 4 + ky - 1, ix = ox * 4 + kx - 1;
            if (iy >= 0 && iy < H && ix >= 0 && ix < W)
                v = im[(c * H + iy) * W + ix];
        }
        sA[e] = (_Float16)v;
    }
    __syncthreads();

    int lane = tid & 31, wid = tid >> 5;
    int m = lane & 15, kh = lane >> 4;       // A/B layouts: lane half selects K chunk
    v8f acc[4][2] = {};
    int n0 = (wid * 2 + 0) * 16 + m;
    int n1 = (wid * 2 + 1) * 16 + m;
#pragma unroll
    for (int ks = 0; ks < 5; ks++) {
        int k0 = ks * 32;
        // B frags from LDS: lane holds column n, contiguous K block of 16
        v16h b0 = *(const v16h*)&sB[(size_t)n0 * KPAD + k0 + kh * 16];
        v16h b1 = *(const v16h*)&sB[(size_t)n1 * KPAD + k0 + kh * 16];
#pragma unroll
        for (int mt = 0; mt < 4; mt++) {
            int mrow = mt * 16 + m;
            v8h lo = *(const v8h*)&sA[mrow * KPAD + k0 + kh * 16];
            v8h hi = *(const v8h*)&sA[mrow * KPAD + k0 + kh * 16 + 8];
            v16h a;
#pragma unroll
            for (int i = 0; i < 8; i++) { a[i] = lo[i]; a[i + 8] = hi[i]; }
            acc[mt][0] = __builtin_amdgcn_wmma_f32_16x16x32_f16(false, a, false, b0, (short)0, acc[mt][0], false, false);
            acc[mt][1] = __builtin_amdgcn_wmma_f32_16x16x32_f16(false, a, false, b1, (short)0, acc[mt][1], false, false);
        }
    }
    // bias + ReLU + partial mean pool over all 64 pixels of this tile
    float bias0 = bc[n0], bias1 = bc[n1];
    float s0 = 0.f, s1 = 0.f;
#pragma unroll
    for (int mt = 0; mt < 4; mt++) {
#pragma unroll
        for (int r = 0; r < 8; r++) {
            s0 += fmaxf(acc[mt][0][r] + bias0, 0.f);
            s1 += fmaxf(acc[mt][1][r] + bias1, 0.f);
        }
    }
    atomicAdd(&pooled[(size_t)(pooledBase + img) * COUT + n0], s0 * invN);
    atomicAdd(&pooled[(size_t)(pooledBase + img) * COUT + n1], s1 * invN);
}

// ------------------------------------- WMMA FC: [176,256] x [256,4096] + bf
__global__ void k_fc(const float* __restrict__ pooled, const _Float16* __restrict__ WfH,
                     const float* __restrict__ bf, const float* __restrict__ maskf,
                     float* __restrict__ outFeats, float* __restrict__ outFull) {
    __shared__ __align__(16) _Float16 sA[16 * COUT];
    int mt = blockIdx.x >> 3;     // 0..10
    int ng = blockIdx.x & 7;      // N group of 512
    int tid = threadIdx.x;
    for (int e = tid; e < 16 * COUT; e += 256) {
        int m = e >> 8, k = e & 255;
        sA[e] = (_Float16)pooled[(size_t)(mt * 16 + m) * COUT + k];
    }
    __syncthreads();
    int lane = tid & 31, wid = tid >> 5;
    int m = lane & 15, kh = lane >> 4;
    v8f acc[4] = {};
    int nbase = (ng * 32 + wid * 4) * 16 + m;
#pragma unroll
    for (int ks = 0; ks < 8; ks++) {
        int k0 = ks * 32;
        v8h lo = *(const v8h*)&sA[m * COUT + k0 + kh * 16];
        v8h hi = *(const v8h*)&sA[m * COUT + k0 + kh * 16 + 8];
        v16h a;
#pragma unroll
        for (int i = 0; i < 8; i++) { a[i] = lo[i]; a[i + 8] = hi[i]; }
#pragma unroll
        for (int t = 0; t < 4; t++) {
            int n = nbase + t * 16;
            v16h b = *(const v16h*)(WfH + (size_t)n * COUT + k0 + kh * 16);
            acc[t] = __builtin_amdgcn_wmma_f32_16x16x32_f16(false, a, false, b, (short)0, acc[t], false, false);
        }
    }
#pragma unroll
    for (int t = 0; t < 4; t++) {
        int n = nbase + t * 16;
        float bias = bf[n];
#pragma unroll
        for (int r = 0; r < 8; r++) {
            int gm = mt * 16 + r + 8 * kh;          // C/D layout: M = r + 8*(lane/16)
            float val = acc[t][r] + bias;
            if (gm < BATCH) {
                outFull[(size_t)gm * FDIM + n] = val;
            } else if (gm < BATCH + NCROPS) {
                int q = gm - BATCH;
                outFeats[(size_t)q * FDIM + n] = val * maskf[q];
            }
        }
    }
}

extern "C" void kernel_launch(void* const* d_in, const int* in_sizes, int n_in,
                              void* d_out, int out_size, void* d_ws, size_t ws_size,
                              hipStream_t stream) {
    const float* x      = (const float*)d_in[0];
    const float* boxes  = (const float*)d_in[1];
    const float* scores = (const float*)d_in[2];
    const int*   labels = (const int*)d_in[3];
    const float* Wc     = (const float*)d_in[4];
    const float* bc     = (const float*)d_in[5];
    const float* Wf     = (const float*)d_in[6];
    const float* bf     = (const float*)d_in[7];

    // workspace layout (bytes, all 256B aligned)
    char* ws = (char*)d_ws;
    float*    images = (float*)(ws + 0);                 // 8*3*512*512 f32 = 25165824 B
    float*    crops  = (float*)(ws + 25165824);          // 160*3*64*64 f32 = 7864320 B
    _Float16* Wt     = (_Float16*)(ws + 33030144);       // 256*160 f16     = 81920 B
    _Float16* WfH    = (_Float16*)(ws + 33112064);       // 4096*256 f16    = 2097152 B
    float*    pooled = (float*)(ws + 35209216);          // 176*256 f32     = 180224 B
    float*    maskf  = (float*)(ws + 35389440);          // 160 f32

    // output layout: Objects[8,20,4] | Object_features[8,20,4096] | valid[8,20] | fullframe[8,4096]
    float* outObj   = (float*)d_out;
    float* outFeats = outObj + BATCH * TOPK * 4;
    float* outValid = outFeats + (size_t)NCROPS * FDIM;
    float* outFull  = outValid + BATCH * TOPK;

    k_normalize<<<(BATCH * 3 * IMH * IMW) / 256, 256, 0, stream>>>(x, images, BATCH * 3 * IMH * IMW);
    k_prep_wc<<<(COUT * KPAD + 255) / 256, 256, 0, stream>>>(Wc, Wt);
    k_prep_wf<<<(FDIM * COUT) / 256, 256, 0, stream>>>(Wf, WfH);
    k_zero<<<(MROWS * COUT + 255) / 256, 256, 0, stream>>>(pooled, MROWS * COUT);
    k_topk<<<BATCH, 256, 0, stream>>>(boxes, scores, labels, outObj, outValid, maskf);
    k_roi<<<(NCROPS * 3 * CROP * CROP) / 256, 256, 0, stream>>>(images, outObj, crops);
    // full-frame conv: 8 images, 128x128 out -> 256 64-pixel tiles each
    k_convpool<<<BATCH * 256, 256, 0, stream>>>(images, Wt, bc, pooled,
                                                256, IMH, IMW, 128, 1.f / 16384.f, 0);
    // crop conv: 160 crops, 16x16 out -> 4 64-pixel tiles each
    k_convpool<<<NCROPS * 4, 256, 0, stream>>>(crops, Wt, bc, pooled,
                                               4, CROP, CROP, 16, 1.f / 256.f, BATCH);
    // FC over stacked pooled features [176,256] -> [176,4096]
    k_fc<<<(MROWS / 16) * 8, 256, 0, stream>>>(pooled, WfH, bf, maskf, outFeats, outFull);
}
